// MambaForecaster_16277926052149
// MI455X (gfx1250) — matrix operand order repaired and checked by reading, not compiled, not run on hardware
//
#include <hip/hip_runtime.h>
#include <hip/hip_bf16.h>

#define B_SZ 4
#define L_SZ 2048
#define DM   256      // D_MODEL
#define DI   512      // D_INNER
#define DS   16       // D_STATE
#define DTR  16       // DT_RANK
#define HZ   96       // HORIZON
#define NROW (B_SZ * L_SZ)   // 8192

typedef __attribute__((ext_vector_type(16))) __bf16      v16bf;
typedef __attribute__((ext_vector_type(8)))  float       v8f;
typedef __attribute__((ext_vector_type(4)))  unsigned int u32x4;

union FragBF { v16bf v; u32x4 q[2]; };

__device__ __forceinline__ float sigmoidf_(float x) { return 1.0f / (1.0f + __expf(-x)); }
__device__ __forceinline__ float softplusf_(float x) {
  return (x > 20.0f) ? x : __logf(1.0f + __expf(x));
}

// async global->LDS copy of 16 bytes (gfx1250, tracked by ASYNCcnt)
__device__ __forceinline__ void async_ld_b128(unsigned lds_off, const void* gaddr) {
  asm volatile("global_load_async_to_lds_b128 %0, %1, off"
               :: "v"(lds_off), "v"(gaddr) : "memory");
}
__device__ __forceinline__ void wait_async0() {
#if __has_builtin(__builtin_amdgcn_s_wait_asynccnt)
  __builtin_amdgcn_s_wait_asynccnt(0);
#else
  asm volatile("s_wait_asynccnt 0x0" ::: "memory");
#endif
}

// ---------------------------------------------------------------- f32 -> bf16 (RNE)
__global__ void cvt_f32_bf16(const float* __restrict__ in, unsigned short* __restrict__ out, int n) {
  int i = blockIdx.x * blockDim.x + threadIdx.x;
  if (i < n) {
    unsigned int b = __float_as_uint(in[i]);
    unsigned int r = (b + 0x7FFFu + ((b >> 16) & 1u)) >> 16;
    out[i] = (unsigned short)r;
  }
}

// ---------------------------------------------------------------- in_proj GEMM (WMMA bf16, async double-buffered LDS)
// xz[8192,1024] = Xb[8192,256] * Wb[1024,256]^T
// Workgroup (256 thr = 8 waves) -> 128x128 tile; K stepped by 32, double-buffered in LDS.
__global__ __launch_bounds__(256) void in_proj_wmma(
    const unsigned short* __restrict__ Xb,
    const unsigned short* __restrict__ Wb,
    float* __restrict__ xz)
{
  __shared__ char smem[32768];       // A0(8K) A1(8K) B0(8K) B1(8K)
  const unsigned smem_base = (unsigned)(unsigned long long)(void*)smem;

  const int tid  = threadIdx.x;
  const int lane = tid & 31;
  const int wave = tid >> 5;
  const int hi   = lane >> 4;
  const int l15  = lane & 15;

  const int tm = blockIdx.x >> 3;    // 0..63  (M tiles of 128)
  const int tn = blockIdx.x & 7;     // 0..7   (N tiles of 128)
  const int wm = wave & 3;           // 4 waves along M -> 32 rows each
  const int wn = wave >> 1 >> 1;     // wave>>2: 2 waves along N -> 64 cols each
  const int m0 = wm * 32;
  const int n0 = wn * 64;

  // LDS tile fill: A/B each 128 rows x 32 bf16 cols = 512 x 16B chunks; 2 chunks/thread/matrix
  auto fill = [&](int kb, int buf) {
    const unsigned ldsA = smem_base + buf * 8192;
    const unsigned ldsB = smem_base + 16384 + buf * 8192;
    #pragma unroll
    for (int s = 0; s < 2; ++s) {
      int ci = tid + s * 256;
      int r = ci >> 2, c = ci & 3;                  // row 0..127, 16B chunk 0..3
      const char* ga = (const char*)Xb + ((size_t)(tm * 128 + r) * DM + kb) * 2 + c * 16;
      async_ld_b128(ldsA + r * 64 + c * 16, ga);
      const char* gb = (const char*)Wb + ((size_t)(tn * 128 + r) * DM + kb) * 2 + c * 16;
      async_ld_b128(ldsB + r * 64 + c * 16, gb);
    }
  };

  v8f acc[2][4];
  #pragma unroll
  for (int f = 0; f < 2; ++f)
    #pragma unroll
    for (int j = 0; j < 4; ++j) acc[f][j] = (v8f){};

  fill(0, 0);
  wait_async0();
  __syncthreads();

  for (int ks = 0; ks < 8; ++ks) {
    if (ks < 7) fill((ks + 1) * 32, (ks + 1) & 1);   // prefetch next K-slab into other buffer

    const unsigned short* Ab = (const unsigned short*)(smem + ((ks & 1) ? 8192 : 0));
    const unsigned short* Bb = (const unsigned short*)(smem + 16384 + ((ks & 1) ? 8192 : 0));

    // A fragments (ISA 16-bit 16x32 layout): lane row = l15, K chunks {hi*8..+7},{16+hi*8..+7}
    FragBF a0, a1;
    a0.q[0] = *(const u32x4*)(Ab + (m0 +      l15) * 32 + hi * 8);
    a0.q[1] = *(const u32x4*)(Ab + (m0 +      l15) * 32 + hi * 8 + 16);
    a1.q[0] = *(const u32x4*)(Ab + (m0 + 16 + l15) * 32 + hi * 8);
    a1.q[1] = *(const u32x4*)(Ab + (m0 + 16 + l15) * 32 + hi * 8 + 16);

    #pragma unroll
    for (int j = 0; j < 4; ++j) {
      // B fragment: lane col = l15, K chunks {hi*8..+7},{16+hi*8..+7}
      FragBF fb;
      fb.q[0] = *(const u32x4*)(Bb + (n0 + j * 16 + l15) * 32 + hi * 8);
      fb.q[1] = *(const u32x4*)(Bb + (n0 + j * 16 + l15) * 32 + hi * 8 + 16);
      acc[0][j] = __builtin_amdgcn_wmma_f32_16x16x32_bf16(false, a0.v, false, fb.v, (short)0, acc[0][j], false, false);
      acc[1][j] = __builtin_amdgcn_wmma_f32_16x16x32_bf16(false, a1.v, false, fb.v, (short)0, acc[1][j], false, false);
    }

    wait_async0();        // next buffer landed
    __syncthreads();      // everyone done reading current buffer / data visible
  }

  // C layout: VGPR r -> M = r + hi*8 within 16-row frag; N = l15
  #pragma unroll
  for (int f = 0; f < 2; ++f) {
    const int mrow = tm * 128 + m0 + f * 16 + hi * 8;
    #pragma unroll
    for (int j = 0; j < 4; ++j) {
      const int ncol = tn * 128 + n0 + j * 16 + l15;
      #pragma unroll
      for (int r = 0; r < 8; ++r)
        xz[(size_t)(mrow + r) * 1024 + ncol] = acc[f][j][r];
    }
  }
}

// ---------------------------------------------------------------- depthwise causal conv(4) + SiLU
__global__ __launch_bounds__(256) void conv_silu(
    const float* __restrict__ xz,      // [NROW,1024], x_in = [:,:512]
    const float* __restrict__ conv_w,  // [512,1,4]
    const float* __restrict__ conv_b,  // [512]
    float* __restrict__ u)             // [NROW,512]
{
  int gid = blockIdx.x * blockDim.x + threadIdx.x;
  if (gid >= NROW * DI) return;
  int d = gid % DI;
  int t = (gid / DI) % L_SZ;
  int b = gid / (DI * L_SZ);
  float acc = conv_b[d];
  #pragma unroll
  for (int j = 0; j < 4; ++j) {
    int tq = t - 3 + j;
    float xv = (tq >= 0) ? xz[((size_t)(b * L_SZ + tq)) * 1024 + d] : 0.0f;
    acc += conv_w[d * 4 + j] * xv;
  }
  u[gid] = acc * sigmoidf_(acc);      // SiLU
}

// ---------------------------------------------------------------- x_proj + dt_proj + softplus (fused, block per row)
__global__ __launch_bounds__(256) void xproj_delta(
    const float* __restrict__ u,          // [NROW,512]
    const float* __restrict__ x_proj_w,   // [48,512]
    const float* __restrict__ dt_proj_w,  // [512,16]
    const float* __restrict__ dt_proj_b,  // [512]
    float* __restrict__ delta,            // [NROW,512]
    float* __restrict__ Bm,               // [NROW,16]
    float* __restrict__ Cm)               // [NROW,16]
{
  __shared__ float su[DI];
  __shared__ float sdt[DTR];
  const int row = blockIdx.x;
  const int tid = threadIdx.x;

  for (int i = tid; i < DI; i += 256) su[i] = u[(size_t)row * DI + i];
  __syncthreads();

  if (tid < 48) {
    const float* wp = x_proj_w + tid * DI;
    float acc = 0.0f;
    for (int k = 0; k < DI; ++k) acc += su[k] * wp[k];
    if (tid < DTR)       sdt[tid] = acc;
    else if (tid < 32)   Bm[(size_t)row * DS + (tid - 16)] = acc;
    else                 Cm[(size_t)row * DS + (tid - 32)] = acc;
  }
  __syncthreads();

  for (int d = tid; d < DI; d += 256) {
    float acc = dt_proj_b[d];
    const float* wp = dt_proj_w + d * DTR;
    #pragma unroll
    for (int r = 0; r < DTR; ++r) acc += sdt[r] * wp[r];
    delta[(size_t)row * DI + d] = softplusf_(acc);
  }
}

// ---------------------------------------------------------------- selective scan (one thread per (b,d,n))
__global__ __launch_bounds__(128) void ssm_scan(
    const float* __restrict__ delta,  // [NROW,512]
    const float* __restrict__ u,      // [NROW,512]
    const float* __restrict__ Bm,     // [NROW,16]
    const float* __restrict__ A_log,  // [512,16]
    float* __restrict__ hout)         // [B,512,16]
{
  __shared__ float sdel[32][8];
  __shared__ float su_[32][8];
  __shared__ float sB[32][17];

  const int b  = blockIdx.x >> 6;       // 0..3
  const int dg = blockIdx.x & 63;       // 0..63
  const int d0 = dg * 8;
  const int tid = threadIdx.x;
  const int dl = tid >> 4;              // 0..7
  const int n  = tid & 15;              // 0..15
  const int d  = d0 + dl;

  const float a_dn = -__expf(A_log[d * DS + n]);   // A = -exp(A_log)
  float h = 0.0f;

  for (int c = 0; c < L_SZ / 32; ++c) {
    const int t0 = c * 32;
    for (int i = tid; i < 256; i += 128) {
      int tt = i >> 3, j = i & 7;
      size_t row = (size_t)(b * L_SZ + t0 + tt);
      sdel[tt][j] = delta[row * DI + d0 + j];
      su_[tt][j]  = u[row * DI + d0 + j];
    }
    for (int i = tid; i < 512; i += 128) {
      int tt = i >> 4, j = i & 15;
      sB[tt][j] = Bm[(size_t)(b * L_SZ + t0 + tt) * DS + j];
    }
    __syncthreads();

    // prefetch next chunk (global_prefetch_b8 on gfx1250)
    if (c + 1 < L_SZ / 32) {
      int t0n = t0 + 32;
      for (int i = tid; i < 32; i += 128) {
        size_t row = (size_t)(b * L_SZ + t0n + i);
        __builtin_prefetch(&delta[row * DI + d0], 0, 1);
        __builtin_prefetch(&u[row * DI + d0], 0, 1);
        __builtin_prefetch(&Bm[row * DS], 0, 1);
      }
    }

    #pragma unroll 8
    for (int tt = 0; tt < 32; ++tt) {
      float dv = sdel[tt][dl];
      float dA = __expf(dv * a_dn);
      h = dA * h + dv * sB[tt][n] * su_[tt][dl];
    }
    __syncthreads();
  }
  hout[((size_t)(b * DI + d)) * DS + n] = h;
}

// ---------------------------------------------------------------- final: y_last -> out_proj -> head (1 block)
__global__ __launch_bounds__(256) void final_head(
    const float* __restrict__ h,       // [B,512,16]
    const float* __restrict__ Cm,      // [NROW,16]
    const float* __restrict__ u,       // [NROW,512]
    const float* __restrict__ xz,      // [NROW,1024] (z = [:,512:])
    const float* __restrict__ Dp,      // [512]
    const float* __restrict__ out_w,   // [256,512]
    const float* __restrict__ head_w,  // [96,256]
    const float* __restrict__ head_b,  // [96]
    float* __restrict__ res)           // [B,96]
{
  __shared__ float sy[DI];
  __shared__ float so[DM];
  const int tid = threadIdx.x;

  for (int b = 0; b < B_SZ; ++b) {
    const size_t lastrow = (size_t)(b * L_SZ + (L_SZ - 1));
    for (int d = tid; d < DI; d += 256) {
      const float* hp = h + ((size_t)(b * DI + d)) * DS;
      const float* cp = Cm + lastrow * DS;
      float acc = 0.0f;
      #pragma unroll
      for (int nn = 0; nn < DS; ++nn) acc += hp[nn] * cp[nn];
      float y = acc + u[lastrow * DI + d] * Dp[d];
      float z = xz[lastrow * 1024 + DI + d];
      sy[d] = y * (z * sigmoidf_(z));
    }
    __syncthreads();
    for (int cc = tid; cc < DM; cc += 256) {
      const float* wp = out_w + cc * DI;
      float acc = 0.0f;
      for (int dd = 0; dd < DI; ++dd) acc += sy[dd] * wp[dd];
      so[cc] = acc;
    }
    __syncthreads();
    for (int hh = tid; hh < HZ; hh += 256) {
      const float* wp = head_w + hh * DM;
      float acc = head_b[hh];
      for (int cc = 0; cc < DM; ++cc) acc += so[cc] * wp[cc];
      res[b * HZ + hh] = acc;
    }
    __syncthreads();
  }
}

// ----------------------------------------------------------------
extern "C" void kernel_launch(void* const* d_in, const int* in_sizes, int n_in,
                              void* d_out, int out_size, void* d_ws, size_t ws_size,
                              hipStream_t stream) {
  const float* x          = (const float*)d_in[0];
  const float* in_proj_w  = (const float*)d_in[1];
  const float* conv_w     = (const float*)d_in[2];
  const float* conv_b     = (const float*)d_in[3];
  const float* x_proj_w   = (const float*)d_in[4];
  const float* dt_proj_w  = (const float*)d_in[5];
  const float* dt_proj_b  = (const float*)d_in[6];
  const float* A_log      = (const float*)d_in[7];
  const float* D_param    = (const float*)d_in[8];
  const float* out_proj_w = (const float*)d_in[9];
  const float* head_w     = (const float*)d_in[10];
  const float* head_b     = (const float*)d_in[11];
  float* res              = (float*)d_out;

  char* p = (char*)d_ws;
  auto alloc = [&](size_t bytes) -> char* {
    char* r = p; p += (bytes + 255) & ~(size_t)255; return r;
  };
  float* xz          = (float*)alloc((size_t)NROW * 1024 * 4);   // 33.5 MB
  float* u           = (float*)alloc((size_t)NROW * DI * 4);     // 16.8 MB
  float* delta       = (float*)alloc((size_t)NROW * DI * 4);     // 16.8 MB
  float* Bmat        = (float*)alloc((size_t)NROW * DS * 4);
  float* Cmat        = (float*)alloc((size_t)NROW * DS * 4);
  float* hfin        = (float*)alloc((size_t)B_SZ * DI * DS * 4);
  unsigned short* Xb = (unsigned short*)alloc((size_t)NROW * DM * 2);
  unsigned short* Wb = (unsigned short*)alloc((size_t)1024 * DM * 2);

  // 1) bf16 copies of X and in_proj_w
  {
    int n = NROW * DM;
    cvt_f32_bf16<<<(n + 255) / 256, 256, 0, stream>>>(x, Xb, n);
    int nw = 1024 * DM;
    cvt_f32_bf16<<<(nw + 255) / 256, 256, 0, stream>>>(in_proj_w, Wb, nw);
  }
  // 2) in_proj GEMM -> xz  (512 WGs, 128x128 tile each, async LDS double-buffer)
  in_proj_wmma<<<512, 256, 0, stream>>>(Xb, Wb, xz);
  // 3) depthwise conv + SiLU -> u
  conv_silu<<<(NROW * DI + 255) / 256, 256, 0, stream>>>(xz, conv_w, conv_b, u);
  // 4) x_proj + dt_proj + softplus -> delta, Bm, Cm
  xproj_delta<<<NROW, 256, 0, stream>>>(u, x_proj_w, dt_proj_w, dt_proj_b, delta, Bmat, Cmat);
  // 5) selective scan -> h_final
  ssm_scan<<<256, 128, 0, stream>>>(delta, u, Bmat, A_log, hfin);
  // 6) last-step epilogue + out_proj + head
  final_head<<<1, 256, 0, stream>>>(hfin, Cmat, u, xz, D_param, out_proj_w, head_w, head_b, res);
}
